// MultiHeadAttention_8924942041424
// MI455X (gfx1250) — compile-verified, matrix operand order checked by reading, not run on hardware
//
#include <hip/hip_runtime.h>
#include <hip/hip_bf16.h>
#include <math.h>

// ---------------------------------------------------------------------------
// Problem constants (reference: B=2, L=2048, D=1024, H=16, d_k=64)
// ---------------------------------------------------------------------------
#define BB 2
#define LL 2048
#define DD 1024
#define HH 16
#define DK 64

typedef __bf16 bf16_t;
typedef __attribute__((ext_vector_type(16))) __bf16 v16bf;
typedef __attribute__((ext_vector_type(8)))  float  v8f;
typedef __attribute__((ext_vector_type(4)))  unsigned int u32x4;

union FragAB { v16bf v; u32x4 u[2]; };

// ---------------------------------------------------------------------------
// WMMA wrapper: D = A(16x32 bf16) x B(32x16 bf16) + C(16x16 f32)
// ---------------------------------------------------------------------------
__device__ __forceinline__ v8f wmma_bf16(v16bf a, v16bf b, v8f c) {
  return __builtin_amdgcn_wmma_f32_16x16x32_bf16(
      /*neg_a=*/false, a, /*neg_b=*/false, b,
      /*c_mod=*/(short)0, c, /*reuse_a=*/false, /*reuse_b=*/false);
}

// ---------------------------------------------------------------------------
// Load an A- or B-fragment for the NT layout.
// A (16x32, MxK): lane = row m (lane&15), kh = lane>>4;
//   elems[0..7]  = M[m][kh*8 + 0..7]
//   elems[8..15] = M[m][16 + kh*8 + 0..7]
// B (32x16, KxN) with W stored (N,K) row-major: identical addressing with
//   "row" = output column n.  ld must be a multiple of 8 elements (16B align).
// ---------------------------------------------------------------------------
__device__ __forceinline__ v16bf load_ab_frag(const bf16_t* __restrict__ p,
                                              int ld) {
  const int lane = threadIdx.x & 31;
  const int r  = lane & 15;
  const int kh = lane >> 4;
  const bf16_t* q = p + (size_t)r * ld + kh * 8;
  FragAB f;
  f.u[0] = *(const u32x4*)(q);
  f.u[1] = *(const u32x4*)(q + 16);
  return f.v;
}

// ---------------------------------------------------------------------------
// fp32 -> bf16 cast
// ---------------------------------------------------------------------------
__global__ void __launch_bounds__(256, 1)
cast_f32_bf16_kernel(const float* __restrict__ in,
                     bf16_t* __restrict__ out, int n) {
  for (int i = blockIdx.x * blockDim.x + threadIdx.x; i < n;
       i += gridDim.x * blockDim.x)
    out[i] = (bf16_t)in[i];
}

// ---------------------------------------------------------------------------
// NT GEMM: C[M,N] (f32) = A[M,K] (bf16, row major) x W[N,K]^T (bf16, row major)
// Block = 256 threads = 8 waves (2 x 4), block tile 128(M) x 256(N),
// each wave computes a 64x64 tile with 4x4 WMMA fragments, K step 32.
// __launch_bounds__(256, 1): the 4x4 f32 accumulator tile (128 VGPRs) plus
// 8 bf16 fragments (64 VGPRs) must be register-resident — without this the
// compiler spills accumulators to scratch inside the K loop.
// ---------------------------------------------------------------------------
__global__ void __launch_bounds__(256, 1)
gemm_nt_bf16_kernel(const bf16_t* __restrict__ A,
                    const bf16_t* __restrict__ W,
                    float* __restrict__ C,
                    int K, int N) {
  const int lane = threadIdx.x & 31;
  const int wv   = threadIdx.x >> 5;      // 0..7
  const int wM   = wv >> 2;               // 0..1
  const int wN   = wv & 3;                // 0..3
  const int n    = lane & 15;
  const int lg   = lane >> 4;

  const int m_base = blockIdx.y * 128 + wM * 64;
  const int n_base = blockIdx.x * 256 + wN * 64;

  v8f acc[4][4];
#pragma unroll
  for (int i = 0; i < 4; ++i)
#pragma unroll
    for (int j = 0; j < 4; ++j) acc[i][j] = (v8f)(0.0f);

  for (int k0 = 0; k0 < K; k0 += 32) {
    v16bf a[4], b[4];
#pragma unroll
    for (int i = 0; i < 4; ++i)
      a[i] = load_ab_frag(A + (size_t)(m_base + 16 * i) * K + k0, K);
#pragma unroll
    for (int j = 0; j < 4; ++j)
      b[j] = load_ab_frag(W + (size_t)(n_base + 16 * j) * K + k0, K);

    if (k0 + 64 < K) {  // hint next K tile into cache (global_prefetch_b8)
      __builtin_prefetch(A + (size_t)m_base * K + k0 + 64, 0, 0);
      __builtin_prefetch(W + (size_t)n_base * K + k0 + 64, 0, 0);
    }

#pragma unroll
    for (int i = 0; i < 4; ++i)
#pragma unroll
      for (int j = 0; j < 4; ++j)
        acc[i][j] = wmma_bf16(a[i], b[j], acc[i][j]);
  }

  // C fragment mapping: element (m, n): lane = n + 16*(m>>3), vgpr = m&7
#pragma unroll
  for (int i = 0; i < 4; ++i)
#pragma unroll
    for (int j = 0; j < 4; ++j) {
      const int col = n_base + 16 * j + n;
#pragma unroll
      for (int r = 0; r < 8; ++r) {
        const int row = m_base + 16 * i + lg * 8 + r;
        C[(size_t)row * N + col] = acc[i][j][r];
      }
    }
}

// ---------------------------------------------------------------------------
// RoPE + layout shuffle.
// qkv (f32) layout: [b, l, 3, H, DK]  (row e = s*D + h*64 + d)
// Writes:  Qr [b,h,l,d] bf16 (rotated, pre-scaled by 1/sqrt(dk))
//          Kr [b,h,l,d] bf16 (rotated)
//          Vt [b,h,d,l] bf16 (transposed for contiguous B-fragment columns)
// One thread per (b,h,l).
// ---------------------------------------------------------------------------
__global__ void __launch_bounds__(256, 1)
rope_kernel(const float* __restrict__ qkv,
            const int* __restrict__ pos,
            bf16_t* __restrict__ Qr,
            bf16_t* __restrict__ Kr,
            bf16_t* __restrict__ Vt) {
  const int idx = blockIdx.x * blockDim.x + threadIdx.x;  // b*H*L threads
  const int l  = idx & (LL - 1);
  const int h  = (idx / LL) & (HH - 1);
  const int b  = idx / (LL * HH);
  if (b >= BB) return;

  const size_t row = ((size_t)(b * LL + l)) * (3 * DD);
  const float* q = qkv + row + 0 * DD + h * DK;
  const float* k = qkv + row + 1 * DD + h * DK;
  const float* v = qkv + row + 2 * DD + h * DK;

  const int bh = b * HH + h;
  bf16_t* qo = Qr + ((size_t)bh * LL + l) * DK;
  bf16_t* ko = Kr + ((size_t)bh * LL + l) * DK;
  bf16_t* vo = Vt + ((size_t)bh * DK) * LL + l;  // stride LL per d

  const float p = (float)pos[l];
  const float scale = 0.125f;  // 1/sqrt(64), folded into Q
  const float LN_THETA = 9.210340371976184f;  // ln(10000)

#pragma unroll
  for (int i = 0; i < DK / 2; ++i) {
    const float inv_freq = __expf(-((float)(2 * i) / (float)DK) * LN_THETA);
    const float ang = p * inv_freq;
    const float c = __cosf(ang), s = __sinf(ang);

    const float qe = q[2 * i], qn = q[2 * i + 1];
    qo[2 * i]     = (bf16_t)((qe * c - qn * s) * scale);
    qo[2 * i + 1] = (bf16_t)((qe * s + qn * c) * scale);

    const float ke = k[2 * i], kn = k[2 * i + 1];
    ko[2 * i]     = (bf16_t)(ke * c - kn * s);
    ko[2 * i + 1] = (bf16_t)(ke * s + kn * c);

    vo[(size_t)(2 * i) * LL]     = (bf16_t)v[2 * i];
    vo[(size_t)(2 * i + 1) * LL] = (bf16_t)v[2 * i + 1];
  }
}

// ---------------------------------------------------------------------------
// Causal flash attention, one wave per 16-query tile, key tiles of 32.
// grid.x = B*H, grid.y = L/(16*8); block = 256 (8 independent waves).
// O written bf16 at (B, L, D) so it feeds the final projection GEMM directly.
// ---------------------------------------------------------------------------
__global__ void __launch_bounds__(256, 1)
attn_kernel(const bf16_t* __restrict__ Qr,
            const bf16_t* __restrict__ Kr,
            const bf16_t* __restrict__ Vt,
            bf16_t* __restrict__ O) {
  __shared__ __attribute__((aligned(16))) bf16_t Plds[8][16][32];  // 8 KB

  const int lane = threadIdx.x & 31;
  const int wv   = threadIdx.x >> 5;
  const int n    = lane & 15;
  const int lg   = lane >> 4;

  const int bh = blockIdx.x;          // b*H + h
  const int b  = bh / HH;
  const int h  = bh % HH;
  const int qt = blockIdx.y * 8 + wv; // 0 .. L/16-1
  const int qbase = qt * 16;

  const bf16_t* Qh  = Qr + ((size_t)bh * LL) * DK;
  const bf16_t* Kh  = Kr + ((size_t)bh * LL) * DK;
  const bf16_t* Vth = Vt + ((size_t)bh * DK) * LL;

  // Q tile fragments (fixed for the whole key loop)
  const v16bf aq0 = load_ab_frag(Qh + (size_t)qbase * DK, DK);       // d 0..31
  const v16bf aq1 = load_ab_frag(Qh + (size_t)qbase * DK + 32, DK);  // d 32..63

  v8f o[4];
#pragma unroll
  for (int nt = 0; nt < 4; ++nt) o[nt] = (v8f)(0.0f);
  float m_r[8], l_r[8];
#pragma unroll
  for (int r = 0; r < 8; ++r) { m_r[r] = -__builtin_inff(); l_r[r] = 0.0f; }

  for (int j0 = 0; j0 <= qbase + 15; j0 += 32) {
    // ---- scores: S = Q K^T (scale folded into Q) -------------------------
    v8f s0 = (v8f)(0.0f), s1 = (v8f)(0.0f);
    v16bf bk;
    bk = load_ab_frag(Kh + (size_t)j0 * DK, DK);             s0 = wmma_bf16(aq0, bk, s0);
    bk = load_ab_frag(Kh + (size_t)j0 * DK + 32, DK);        s0 = wmma_bf16(aq1, bk, s0);
    bk = load_ab_frag(Kh + (size_t)(j0 + 16) * DK, DK);      s1 = wmma_bf16(aq0, bk, s1);
    bk = load_ab_frag(Kh + (size_t)(j0 + 16) * DK + 32, DK); s1 = wmma_bf16(aq1, bk, s1);

    // ---- causal mask + online softmax ------------------------------------
#pragma unroll
    for (int r = 0; r < 8; ++r) {
      const int qg = qbase + lg * 8 + r;
      if (j0 + n > qg)      s0[r] = -__builtin_inff();
      if (j0 + 16 + n > qg) s1[r] = -__builtin_inff();
    }
#pragma unroll
    for (int r = 0; r < 8; ++r) {
      float mx = fmaxf(s0[r], s1[r]);
#pragma unroll
      for (int off = 1; off < 16; off <<= 1)
        mx = fmaxf(mx, __shfl_xor(mx, off, 32));
      const float mn = fmaxf(m_r[r], mx);
      const float alpha = __expf(m_r[r] - mn);  // 0 on first tile
      m_r[r] = mn;
      const float p0 = __expf(s0[r] - mn);
      const float p1 = __expf(s1[r] - mn);
      s0[r] = p0; s1[r] = p1;
      float ps = p0 + p1;
#pragma unroll
      for (int off = 1; off < 16; off <<= 1) ps += __shfl_xor(ps, off, 32);
      l_r[r] = l_r[r] * alpha + ps;
#pragma unroll
      for (int nt = 0; nt < 4; ++nt) o[nt][r] *= alpha;
    }

    // ---- transpose P (score layout -> A layout) through per-wave LDS -----
#pragma unroll
    for (int r = 0; r < 8; ++r) {
      Plds[wv][lg * 8 + r][n]      = (bf16_t)s0[r];
      Plds[wv][lg * 8 + r][n + 16] = (bf16_t)s1[r];
    }
    __builtin_amdgcn_fence(__ATOMIC_ACQ_REL, "wavefront");
    FragAB pf;
    pf.u[0] = *(const u32x4*)&Plds[wv][n][lg * 8];
    pf.u[1] = *(const u32x4*)&Plds[wv][n][16 + lg * 8];
    const v16bf pA = pf.v;
    __builtin_amdgcn_fence(__ATOMIC_ACQ_REL, "wavefront");

    // ---- O += P V  (V^T gives contiguous B-fragment columns) -------------
#pragma unroll
    for (int nt = 0; nt < 4; ++nt) {
      const v16bf bv = load_ab_frag(Vth + (size_t)(nt * 16) * LL + j0, LL);
      o[nt] = wmma_bf16(pA, bv, o[nt]);
    }
  }

  // ---- normalize and emit (B, L, D) bf16 ---------------------------------
  float inv_l[8];
#pragma unroll
  for (int r = 0; r < 8; ++r) inv_l[r] = 1.0f / l_r[r];
#pragma unroll
  for (int nt = 0; nt < 4; ++nt) {
    const int col = h * DK + nt * 16 + n;
#pragma unroll
    for (int r = 0; r < 8; ++r) {
      const int row = qbase + lg * 8 + r;
      O[((size_t)b * LL + row) * DD + col] = (bf16_t)(o[nt][r] * inv_l[r]);
    }
  }
}

// ---------------------------------------------------------------------------
// Host-side launch
// ---------------------------------------------------------------------------
extern "C" void kernel_launch(void* const* d_in, const int* in_sizes, int n_in,
                              void* d_out, int out_size, void* d_ws,
                              size_t ws_size, hipStream_t stream) {
  (void)in_sizes; (void)n_in; (void)out_size; (void)ws_size;

  const float* x    = (const float*)d_in[0];   // (B, L, D)
  const float* Wqkv = (const float*)d_in[1];   // (3D, D)
  const float* Wout = (const float*)d_in[2];   // (D, D)
  const int*   pos  = (const int*)d_in[3];     // (L,)

  const int ML = BB * LL;  // 4096 rows

  char* ws = (char*)d_ws;
  bf16_t* xb    = (bf16_t*)(ws);                          //  8 MB
  bf16_t* wqkvb = (bf16_t*)(ws + (size_t)8  * (1u << 20));//  6 MB
  bf16_t* woutb = (bf16_t*)(ws + (size_t)14 * (1u << 20));//  2 MB
  float*  qkv   = (float*) (ws + (size_t)16 * (1u << 20));// 48 MB
  bf16_t* Qr    = (bf16_t*)(ws + (size_t)64 * (1u << 20));//  8 MB
  bf16_t* Kr    = (bf16_t*)(ws + (size_t)72 * (1u << 20));//  8 MB
  bf16_t* Vt    = (bf16_t*)(ws + (size_t)80 * (1u << 20));//  8 MB
  bf16_t* Obuf  = (bf16_t*)(ws + (size_t)88 * (1u << 20));//  8 MB

  // 1. fp32 -> bf16 conversions
  {
    int n = ML * DD;
    cast_f32_bf16_kernel<<<(n + 255) / 256, 256, 0, stream>>>(x, xb, n);
    n = 3 * DD * DD;
    cast_f32_bf16_kernel<<<(n + 255) / 256, 256, 0, stream>>>(Wqkv, wqkvb, n);
    n = DD * DD;
    cast_f32_bf16_kernel<<<(n + 255) / 256, 256, 0, stream>>>(Wout, woutb, n);
  }

  // 2. QKV projection: (4096 x 1024) x (3072 x 1024)^T -> f32
  gemm_nt_bf16_kernel<<<dim3((3 * DD) / 256, ML / 128), 256, 0, stream>>>(
      xb, wqkvb, qkv, DD, 3 * DD);

  // 3. RoPE + reshape to per-head layouts (V transposed)
  rope_kernel<<<(BB * HH * LL) / 256, 256, 0, stream>>>(qkv, pos, Qr, Kr, Vt);

  // 4. causal flash attention
  attn_kernel<<<dim3(BB * HH, LL / 128), 256, 0, stream>>>(Qr, Kr, Vt, Obuf);

  // 5. output projection: (4096 x 1024) x (1024 x 1024)^T -> d_out (f32)
  gemm_nt_bf16_kernel<<<dim3(DD / 256, ML / 128), 256, 0, stream>>>(
      Obuf, woutb, (float*)d_out, DD, DD);
}
